// ScaledDotProductAttention_36421322670493
// MI455X (gfx1250) — matrix operand order repaired and checked by reading, not compiled, not run on hardware
//
#include <hip/hip_runtime.h>
#include <hip/hip_bf16.h>

// ---------------------------------------------------------------------------
// Fused relative-position flash attention for MI455X (gfx1250, wave32).
//   B=8 H=12 L=1024 D=64  (BH = 96 heads)
// Workgroup: 512 threads = 16 waves. blockIdx.x = head-block (6), .y = q-tile (64).
//   wave w : head  hb*16+w   for QK^T / softmax / PV
//   wave w : q-row qt*16+w   for a_scores^T (n=16 heads as WMMA N dim)
// All matmuls on v_wmma_f32_16x16x32_bf16, f32 accumulate.
// Staging is vectorized: b128 global loads -> b64 LDS stores (bf16).
// ---------------------------------------------------------------------------

typedef __attribute__((ext_vector_type(16))) __bf16        v16bf;
typedef __attribute__((ext_vector_type(8)))  float         v8f;
typedef __attribute__((ext_vector_type(4)))  unsigned int  u32x4;

union Frag {
  unsigned int u[8];
  u32x4        q[2];
  v16bf        v;
};

union Pack4 {
  __bf16             h[4];
  unsigned long long u;
};

__device__ __forceinline__ v8f wmma_bf16(v16bf a, v16bf b, v8f c) {
  return __builtin_amdgcn_wmma_f32_16x16x32_bf16(
      /*neg_a=*/false, a, /*neg_b=*/false, b,
      /*c_mod=*/(short)0, c, /*reuse_a=*/false, /*reuse_b=*/false);
}

#define Lc   1024
#define Dc   64
#define Hc   12
#define BHc  96
#define NKc  32            // keys per chunk (= WMMA contraction for PV)
#define NCH  (Lc / NKc)    // 32 chunks

// LDS layout (bytes)
#define KOFF 0                              // kbuf: 16 waves x [32 k][64 d] bf16
#define VOFF (KOFF + 16 * NKc * Dc * 2)     // vbuf: 16 waves x [64 d][32 k] bf16 (transposed)
#define AOFF (VOFF + 16 * Dc * NKc * 2)     // abuf: 16 waves x [32 k][64 d] bf16 (also Q/R staging)
#define SOFF (AOFF + 16 * NKc * Dc * 2)     // sbuf: [16 q][32 k][16 n] f32 score exchange
#define POFF (SOFF + 16 * NKc * 16 * 4)     // pbuf: 16 waves x [16 q][32 k] bf16
#define SMEM_TOTAL (POFF + 16 * 16 * NKc * 2)   // = 245760 bytes (240 KB)

__global__ __launch_bounds__(512, 1)
void relattn_kernel(const float* __restrict__ Q, const float* __restrict__ K,
                    const float* __restrict__ V, const unsigned char* __restrict__ Mk,
                    const float* __restrict__ A, const float* __restrict__ R,
                    float* __restrict__ O) {
  extern __shared__ char smem[];

  const int tid = threadIdx.x;
  const int w   = tid >> 5;          // wave 0..15
  const int ln  = tid & 31;          // lane
  const int lc  = ln & 15;           // column within 16-lane half
  const int hi  = ln >> 4;           // which half of the wave
  const int hb  = blockIdx.x;        // head-block 0..5
  const int qt  = blockIdx.y;        // q-tile 0..63
  const int gh  = hb * 16 + w;       // global head index 0..95
  const int b   = gh / Hc;           // batch for mask
  const int q0  = qt * 16;
  const int qw  = q0 + w;            // this wave's q-row for a_scores

  __bf16* kb = (__bf16*)(smem + KOFF) + w * (NKc * Dc);
  __bf16* vb = (__bf16*)(smem + VOFF) + w * (Dc * NKc);
  __bf16* ab = (__bf16*)(smem + AOFF) + w * (NKc * Dc);
  float*  sb = (float*)(smem + SOFF);
  __bf16* pb = (__bf16*)(smem + POFF) + w * (16 * NKc);

  const float scale = 0.125f;  // 1/sqrt(64)

  // ------------------ stage Q tile (wave-private) -> qfrag ------------------
  {
    const float* Qp = Q + ((size_t)gh * Lc + q0) * Dc;
#pragma unroll
    for (int t = 0; t < 8; ++t) {
      int i4 = (t * 32 + ln) * 4;
      float4 v4 = *(const float4*)(Qp + i4);
      Pack4 p; p.h[0] = (__bf16)v4.x; p.h[1] = (__bf16)v4.y;
               p.h[2] = (__bf16)v4.z; p.h[3] = (__bf16)v4.w;
      *(unsigned long long*)(ab + i4) = p.u;
    }
  }
  Frag qf[2];
#pragma unroll
  for (int f = 0; f < 2; ++f)
#pragma unroll
    for (int v = 0; v < 8; ++v) {
      int d = 32 * f + 2 * v + 16 * hi;     // A-matrix 16x32, contiguous k = 16*hi + s
      qf[f].u[v] = *(const unsigned int*)((const char*)ab + (lc * Dc + d) * 2);
    }

  // ------------------ stage r_q2 tile (wave-private) -> rfrag ---------------
  {
    const float* Rp = R + ((size_t)qw * BHc + hb * 16) * Dc;
#pragma unroll
    for (int t = 0; t < 8; ++t) {
      int i4 = (t * 32 + ln) * 4;
      float4 v4 = *(const float4*)(Rp + i4);
      Pack4 p; p.h[0] = (__bf16)v4.x; p.h[1] = (__bf16)v4.y;
               p.h[2] = (__bf16)v4.z; p.h[3] = (__bf16)v4.w;
      *(unsigned long long*)(ab + i4) = p.u;
    }
  }
  Frag rf[2];
#pragma unroll
  for (int f = 0; f < 2; ++f) {
    const char* base = (const char*)ab + (lc * Dc + 32 * f + 16 * hi) * 2;  // B-matrix 32x16 layout
    rf[f].q[0] = *(const u32x4*)base;
    rf[f].q[1] = *(const u32x4*)(base + 16);
  }

  // ------------------ accumulators & softmax state --------------------------
  v8f acc[4];
  float m[8], l[8];
#pragma unroll
  for (int t = 0; t < 4; ++t)
#pragma unroll
    for (int i = 0; i < 8; ++i) acc[t][i] = 0.0f;
#pragma unroll
  for (int i = 0; i < 8; ++i) { m[i] = -1e30f; l[i] = 0.0f; }

  const float* Kbase = K + (size_t)gh * Lc * Dc;
  const float* Vbase = V + (size_t)gh * Lc * Dc;
  const float* Abase = A + (size_t)qw * Lc * Dc;
  const unsigned char* Mbase = Mk + ((size_t)b * Lc + q0) * Lc;

  // =========================== main k loop =================================
  for (int kc = 0; kc < NCH; ++kc) {
    const int k0 = kc * NKc;
    const float* Kp = Kbase + (size_t)k0 * Dc;
    const float* Vp = Vbase + (size_t)k0 * Dc;
    const float* Ap = Abase + (size_t)k0 * Dc;

    // ---- stage K [k][d] and a [k][d]: b128 loads -> b64 bf16 stores -------
#pragma unroll 4
    for (int t = 0; t < 16; ++t) {
      int i4 = (t * 32 + ln) * 4;                 // 2048 elems / (32 lanes * 4)
      float4 kv = *(const float4*)(Kp + i4);
      float4 av = *(const float4*)(Ap + i4);
      Pack4 kp, ap;
      kp.h[0] = (__bf16)kv.x; kp.h[1] = (__bf16)kv.y;
      kp.h[2] = (__bf16)kv.z; kp.h[3] = (__bf16)kv.w;
      ap.h[0] = (__bf16)av.x; ap.h[1] = (__bf16)av.y;
      ap.h[2] = (__bf16)av.z; ap.h[3] = (__bf16)av.w;
      *(unsigned long long*)(kb + i4) = kp.u;
      *(unsigned long long*)(ab + i4) = ap.u;
    }
    // ---- stage V transposed [d][k]: b64 loads, b16 transpose scatter ------
#pragma unroll 4
    for (int t = 0; t < 32; ++t) {
      int i2 = (t * 32 + ln) * 2;
      float2 vv = *(const float2*)(Vp + i2);
      int kk = i2 >> 6, d = i2 & 63;
      vb[d * NKc + kk]       = (__bf16)vv.x;
      vb[(d + 1) * NKc + kk] = (__bf16)vv.y;
    }
    if (kc + 1 < NCH) {   // pull next chunk toward L2 (global_prefetch_b8)
      __builtin_prefetch(Kp + NKc * Dc, 0, 1);
      __builtin_prefetch(Vp + NKc * Dc, 0, 1);
      __builtin_prefetch(Ap + NKc * Dc, 0, 1);
    }

    // ---- a_scores^T : rows k (2 tiles of 16), cols n = 16 heads -----------
#pragma unroll
    for (int mt = 0; mt < 2; ++mt) {
      Frag af0, af1;
#pragma unroll
      for (int v = 0; v < 8; ++v) {
        int row = mt * 16 + lc;
        int d0  = 2 * v + 16 * hi;
        af0.u[v] = *(const unsigned int*)((const char*)ab + (row * Dc + d0) * 2);
        af1.u[v] = *(const unsigned int*)((const char*)ab + (row * Dc + 32 + d0) * 2);
      }
      v8f c = {};
      c = wmma_bf16(af0.v, rf[0].v, c);
      c = wmma_bf16(af1.v, rf[1].v, c);
#pragma unroll
      for (int i = 0; i < 8; ++i) {       // write to exchange buffer [q][k][n]
        int krow = mt * 16 + i + 8 * hi;
        sb[(w * NKc + krow) * 16 + lc] = c[i];
      }
    }

    // ---- content scores: 2 tiles of [16 q][16 k] --------------------------
    v8f s[2];
#pragma unroll
    for (int nt = 0; nt < 2; ++nt) {
      Frag kf0, kf1;
      const char* baseB = (const char*)kb + ((nt * 16 + lc) * Dc) * 2;
      kf0.q[0] = *(const u32x4*)(baseB + (16 * hi) * 2);
      kf0.q[1] = *(const u32x4*)(baseB + (16 * hi) * 2 + 16);
      kf1.q[0] = *(const u32x4*)(baseB + (32 + 16 * hi) * 2);
      kf1.q[1] = *(const u32x4*)(baseB + (32 + 16 * hi) * 2 + 16);
      v8f c = {};
      c = wmma_bf16(qf[0].v, kf0.v, c);
      c = wmma_bf16(qf[1].v, kf1.v, c);
      s[nt] = c;
    }

    __syncthreads();   // a_scores exchange buffer ready

    // ---- combine + mask + online softmax ----------------------------------
#pragma unroll
    for (int i = 0; i < 8; ++i) {
      int qq = i + 8 * hi;
#pragma unroll
      for (int nt = 0; nt < 2; ++nt) {
        int kk = nt * 16 + lc;
        float sv = (s[nt][i] + sb[(qq * NKc + kk) * 16 + w]) * scale;
        unsigned char mv = Mbase[(size_t)qq * Lc + k0 + kk];
        s[nt][i] = mv ? -1e9f : sv;
      }
    }
#pragma unroll
    for (int i = 0; i < 8; ++i) {
      float mx = fmaxf(s[0][i], s[1][i]);
      mx = fmaxf(mx, __shfl_xor(mx, 1));
      mx = fmaxf(mx, __shfl_xor(mx, 2));
      mx = fmaxf(mx, __shfl_xor(mx, 4));
      mx = fmaxf(mx, __shfl_xor(mx, 8));
      float mcur = fmaxf(m[i], mx);
      float corr = __expf(m[i] - mcur);
      m[i] = mcur;
      float p0 = __expf(s[0][i] - mcur);
      float p1 = __expf(s[1][i] - mcur);
      float rs = p0 + p1;
      rs += __shfl_xor(rs, 1);
      rs += __shfl_xor(rs, 2);
      rs += __shfl_xor(rs, 4);
      rs += __shfl_xor(rs, 8);
      l[i] = l[i] * corr + rs;
#pragma unroll
      for (int t = 0; t < 4; ++t) acc[t][i] *= corr;
      s[0][i] = p0; s[1][i] = p1;
    }

    __syncthreads();   // all sbuf reads done before next chunk overwrites it

    // ---- P (bf16) through pbuf -> A-fragment ------------------------------
#pragma unroll
    for (int nt = 0; nt < 2; ++nt)
#pragma unroll
      for (int i = 0; i < 8; ++i)
        pb[(i + 8 * hi) * NKc + nt * 16 + lc] = (__bf16)s[nt][i];

    Frag pf;
#pragma unroll
    for (int v = 0; v < 8; ++v) {
      int kkk = 2 * v + 16 * hi;
      pf.u[v] = *(const unsigned int*)((const char*)pb + (lc * NKc + kkk) * 2);
    }

    // ---- PV: acc[dt] += P(16x32) x V(32x16) -------------------------------
#pragma unroll
    for (int dt = 0; dt < 4; ++dt) {
      Frag vf;
      const char* vbase = (const char*)vb + ((dt * 16 + lc) * NKc + 16 * hi) * 2;
      vf.q[0] = *(const u32x4*)vbase;
      vf.q[1] = *(const u32x4*)(vbase + 16);
      acc[dt] = wmma_bf16(pf.v, vf.v, acc[dt]);
    }
  }

  // ------------------ finalize: out = acc / l ------------------------------
  float* Op = O + ((size_t)gh * Lc + q0) * Dc;
#pragma unroll
  for (int i = 0; i < 8; ++i) {
    float inv = 1.0f / (l[i] + 1e-20f);
#pragma unroll
    for (int dt = 0; dt < 4; ++dt)
      Op[(size_t)(i + 8 * hi) * Dc + dt * 16 + lc] = acc[dt][i] * inv;
  }
}

extern "C" void kernel_launch(void* const* d_in, const int* in_sizes, int n_in,
                              void* d_out, int out_size, void* d_ws, size_t ws_size,
                              hipStream_t stream) {
  const float*         Q  = (const float*)d_in[0];
  const float*         K  = (const float*)d_in[1];
  const float*         V  = (const float*)d_in[2];
  const unsigned char* Mk = (const unsigned char*)d_in[3];
  const float*         A  = (const float*)d_in[4];
  const float*         R  = (const float*)d_in[5];
  float*               O  = (float*)d_out;

  (void)in_sizes; (void)n_in; (void)out_size; (void)d_ws; (void)ws_size;

  static_assert(SMEM_TOTAL == 245760, "LDS budget");
  hipFuncSetAttribute((const void*)relattn_kernel,
                      hipFuncAttributeMaxDynamicSharedMemorySize, SMEM_TOTAL);

  dim3 grid(BHc / 16, Lc / 16, 1);   // (6, 64): head-blocks fastest -> a-slice L2 reuse
  dim3 block(512, 1, 1);             // 16 waves
  relattn_kernel<<<grid, block, SMEM_TOTAL, stream>>>(Q, K, V, Mk, A, R, O);
}
